// EncoderTreeSpanNN_21844203668322
// MI455X (gfx1250) — matrix operand order repaired and checked by reading, not compiled, not run on hardware
//
#include <hip/hip_runtime.h>

typedef __attribute__((ext_vector_type(16))) _Float16 v16h;
typedef __attribute__((ext_vector_type(8)))  _Float16 v8h;
typedef __attribute__((ext_vector_type(8)))  float    v8f;
typedef _Float16 half_t;

// Problem constants (fixed by reference setup_inputs)
constexpr int VOC  = 50000;
constexpr int D    = 128;
constexpr int HOPS = 3;
constexpr int BATCH= 16;
constexpr int LC   = 256;
constexpr int MC   = 8;
constexpr int LK   = 512;
constexpr int MK   = 8;

constexpr int MTILE = 32;   // Lc rows per block
constexpr int NTILE = 64;   // kb keys per chunk

// ---------------- WMMA fragment loaders (wave32 CDNA5 layouts) ----------------
// All loaders read two aligned 16-byte chunks -> backend emits ds_load_b128.

__device__ inline v16h concat8(v8h lo, v8h hi) {
  v16h r;
#pragma unroll
  for (int i = 0; i < 8; ++i) { r[i] = lo[i]; r[i + 8] = hi[i]; }
  return r;
}

// A fragment 16x32 f16. Source row-major [row][k], lda in halves (multiple of 8).
// Lanes 0-15: M=lane, K = 0..7 (v0-3) and 16..23 (v4-7)
// Lanes 16-31: M=lane-16, K = 8..15 and 24..31
__device__ inline v16h load_a_frag(const half_t* lds, int lda, int row0, int k0, int lane) {
  const int r  = row0 + (lane & 15);
  const int kb = k0 + ((lane & 16) >> 1);      // +8 halves for upper half-wave
  const half_t* base = lds + r * lda + kb;
  v8h lo = *(const v8h*)(base);                // K = kb .. kb+7
  v8h hi = *(const v8h*)(base + 16);           // K = kb+16 .. kb+23
  return concat8(lo, hi);
}

// B fragment 32x16 f16 where the SOURCE is row-major [n][k] (we use its transpose):
// B[k][n] = src[(n0+n)*ld + k0+k].  N=lane%16, K = 0..7,16..23 / 8..15,24..31 by lane half.
__device__ inline v16h load_bT_frag(const half_t* lds, int ld, int n0, int k0, int lane) {
  const int n  = n0 + (lane & 15);
  const int kb = k0 + ((lane & 16) >> 1);      // +8 halves for upper half-wave
  const half_t* base = lds + n * ld + kb;
  v8h lo = *(const v8h*)(base);                // K = kb .. kb+7
  v8h hi = *(const v8h*)(base + 16);           // K = kb+16 .. kb+23
  return concat8(lo, hi);
}

__device__ inline v8f wmma_f16(v16h a, v16h b, v8f c) {
  return __builtin_amdgcn_wmma_f32_16x16x32_f16(false, a, false, b, (short)0, c, false, false);
}

// ---------------- Kernel 1: fused embedding-bag (gather + sum over span) -----
// One wave per output row; lane covers 4 consecutive d's. Tables are L2-resident
// (153MB total vs 192MB L2). Writes half-precision staged features to workspace.
__global__ void embed_bag_kernel(const int* __restrict__ conv_seqs,
                                 const int* __restrict__ kb_arr,
                                 const float* __restrict__ Ctab,
                                 const float* __restrict__ Ktab,
                                 half_t* __restrict__ kf,
                                 half_t* __restrict__ cf) {
  const int wave = (blockIdx.x * blockDim.x + threadIdx.x) >> 5;
  const int lane = threadIdx.x & 31;
  const int KF_ROWS = HOPS * BATCH * LK;
  const int CF_ROWS = HOPS * BATCH * LC;
  if (wave >= KF_ROWS + CF_ROWS) return;

  const int d0 = lane * 4;
  const int*   idxp;
  const float* tab;
  half_t*      outp;
  if (wave < KF_ROWS) {
    const int hop = wave / (BATCH * LK);
    const int rem = wave - hop * (BATCH * LK);   // b*LK + key
    idxp = kb_arr + rem * MK;
    tab  = Ktab + (size_t)hop * VOC * D;
    outp = kf + (size_t)wave * D;
  } else {
    const int w2  = wave - KF_ROWS;
    const int hop = w2 / (BATCH * LC);
    const int rem = w2 - hop * (BATCH * LC);     // b*LC + pos
    idxp = conv_seqs + rem * MC;
    tab  = Ctab + (size_t)hop * VOC * D;
    outp = cf + (size_t)w2 * D;
  }

  float4 acc = make_float4(0.f, 0.f, 0.f, 0.f);
#pragma unroll
  for (int m = 0; m < 8; ++m) {
    const int idx = idxp[m];                     // uniform per wave -> broadcast
    const float4 v = *(const float4*)(tab + (size_t)idx * D + d0);
    acc.x += v.x; acc.y += v.y; acc.z += v.z; acc.w += v.w;
  }
  half_t h[4] = {(half_t)acc.x, (half_t)acc.y, (half_t)acc.z, (half_t)acc.w};
  *(uint2*)(outp + d0) = *(const uint2*)h;
}

// ---------------- Kernel 1b: kf transpose -> kfT[slab][d][key] ---------------
// LDS-tiled 32x32 transpose; both global sides coalesced. 12.6MB total traffic.
__global__ void transpose_kf_kernel(const half_t* __restrict__ in,
                                    half_t* __restrict__ outT) {
  __shared__ half_t tile[32][33];
  const int slab = blockIdx.z;                 // hop*BATCH + b
  const int k0 = blockIdx.x * 32;              // key tile
  const int d0 = blockIdx.y * 32;              // d tile
  const half_t* src = in + (size_t)slab * LK * D;
  half_t* dst = outT + (size_t)slab * D * LK;
  const int tx = threadIdx.x & 31;
  const int ty = threadIdx.x >> 5;             // 0..7
#pragma unroll
  for (int i = 0; i < 32; i += 8)
    tile[ty + i][tx] = src[(size_t)(k0 + ty + i) * D + d0 + tx];
  __syncthreads();
#pragma unroll
  for (int i = 0; i < 32; i += 8)
    dst[(size_t)(d0 + ty + i) * LK + k0 + tx] = tile[tx][ty + i];
}

// ---------------- Kernel 2: flash-attention over KB per (b, Lc-tile) ---------
// 8 waves/block. GEMM1: wave (mt,nt) owns one 16x16 S tile (A=Q, B=kf^T from sK).
// GEMM2: wave owns two 16x16 O d-tiles (A=P, B=kf from sKT, contiguous-K loads).
// Online softmax never materializes the 256x512 score matrix; only 32x64 chunks sit in LDS.
__global__ __launch_bounds__(256) void memnn_attn_kernel(const half_t* __restrict__ kf,
                                                         const half_t* __restrict__ kfT,
                                                         const half_t* __restrict__ cf,
                                                         float* __restrict__ out) {
  __shared__ half_t sQ [MTILE * D];      // 8 KB   [lc][d]
  __shared__ half_t sK [NTILE * D];      // 16 KB  [key][d]
  __shared__ half_t sKT[D * NTILE];      // 16 KB  [d][key]
  __shared__ float  sS [MTILE * NTILE];  // 8 KB
  __shared__ half_t sP [MTILE * NTILE];  // 4 KB
  __shared__ float  sM[MTILE], sL[MTILE], sScale[MTILE];

  const int tid  = threadIdx.x;
  const int lane = tid & 31;
  const int wave = tid >> 5;             // 0..7
  const int b    = blockIdx.y;
  const int lc0  = blockIdx.x * MTILE;
  const int mt   = wave >> 2;            // 0..1 : M tile
  const int nt   = wave & 3;             // 0..3 : N tile (GEMM1) / d-tile pair (GEMM2)
  const int dt0  = nt * 2, dt1 = nt * 2 + 1;

  const int mrow_base = mt * 16 + ((lane & 16) >> 1); // +v gives row in [0,MTILE)
  const int ncol      = lane & 15;

  v8f accO0 = {}; v8f accO1 = {};        // summed over hops

  for (int hop = 0; hop < HOPS; ++hop) {
    // stage Q tile
    const half_t* cfp = cf + ((size_t)(hop * BATCH + b) * LC + lc0) * D;
    for (int i = tid; i < MTILE * D / 8; i += 256)
      ((uint4*)sQ)[i] = ((const uint4*)cfp)[i];
    if (tid < MTILE) { sM[tid] = -1e30f; sL[tid] = 0.f; }
    v8f o0 = {}; v8f o1 = {};
    __syncthreads();

    for (int n0 = 0; n0 < LK; n0 += NTILE) {
      // stage kf chunk in both layouts
      const half_t* kfp  = kf  + ((size_t)(hop * BATCH + b) * LK + n0) * D;
      const half_t* kfTp = kfT + ((size_t)(hop * BATCH + b) * D) * LK + n0;
      for (int i = tid; i < NTILE * D / 8; i += 256)
        ((uint4*)sK)[i] = ((const uint4*)kfp)[i];
      for (int i = tid; i < D * NTILE / 8; i += 256) {
        const int row = i >> 3;          // d
        const int seg = i & 7;           // 8-half segment within the 64-key row
        ((uint4*)sKT)[i] = *(const uint4*)(kfTp + (size_t)row * LK + seg * 8);
      }
      __syncthreads();

      // S = Q @ kf^T over D (4 WMMA k-steps)
      v8f s = {};
#pragma unroll
      for (int kk = 0; kk < D; kk += 32) {
        v16h a  = load_a_frag (sQ, D, mt * 16, kk, lane);
        v16h bt = load_bT_frag(sK, D, nt * 16, kk, lane);
        s = wmma_f16(a, bt, s);
      }
#pragma unroll
      for (int v = 0; v < 8; ++v)
        sS[(mrow_base + v) * NTILE + nt * 16 + ncol] = s[v];
      __syncthreads();

      // online softmax: one lane per row
      if (tid < MTILE) {
        const float* row = sS + tid * NTILE;
        float mOld = sM[tid];
        float mx = mOld;
        for (int j = 0; j < NTILE; ++j) mx = fmaxf(mx, row[j]);
        const float scale = __expf(mOld - mx);
        float sum = 0.f;
        half_t* prow = sP + tid * NTILE;
        for (int j = 0; j < NTILE; ++j) {
          const float p = __expf(row[j] - mx);
          sum += p;
          prow[j] = (half_t)p;
        }
        sM[tid] = mx;
        sL[tid] = sL[tid] * scale + sum;
        sScale[tid] = scale;
      }
      __syncthreads();

      // rescale running O by correction factor
#pragma unroll
      for (int v = 0; v < 8; ++v) {
        const float sc = sScale[mrow_base + v];
        o0[v] *= sc; o1[v] *= sc;
      }
      // O += P @ kf_chunk (contraction over NTILE keys, 2 WMMA k-steps)
      // B[k=key][n=d] = sKT[d][key] -> contiguous along K (transposed-source path)
#pragma unroll
      for (int kk = 0; kk < NTILE; kk += 32) {
        v16h a  = load_a_frag (sP,  NTILE, mt * 16,  kk, lane);
        v16h b0 = load_bT_frag(sKT, NTILE, dt0 * 16, kk, lane);
        o0 = wmma_f16(a, b0, o0);
        v16h b1 = load_bT_frag(sKT, NTILE, dt1 * 16, kk, lane);
        o1 = wmma_f16(a, b1, o1);
      }
      __syncthreads();   // protect sK/sKT/sS/sP before next chunk
    }

    // finalize hop: acc += O / l
#pragma unroll
    for (int v = 0; v < 8; ++v) {
      const float inv = 1.0f / sL[mrow_base + v];
      accO0[v] += o0[v] * inv;
      accO1[v] += o1[v] * inv;
    }
    __syncthreads();     // protect stats/sQ before next hop
  }

  // store out[lc, b, d]
#pragma unroll
  for (int v = 0; v < 8; ++v) {
    const int row = lc0 + mrow_base + v;
    float* op = out + (size_t)row * (BATCH * D) + b * D;
    op[dt0 * 16 + ncol] = accO0[v];
    op[dt1 * 16 + ncol] = accO1[v];
  }
}

// ---------------------------------------------------------------------------
extern "C" void kernel_launch(void* const* d_in, const int* in_sizes, int n_in,
                              void* d_out, int out_size, void* d_ws, size_t ws_size,
                              hipStream_t stream) {
  const int*   conv_seqs = (const int*)d_in[0];
  const int*   kb_arr    = (const int*)d_in[1];
  const float* Ctab      = (const float*)d_in[2];
  const float* Ktab      = (const float*)d_in[3];
  float*       out       = (float*)d_out;

  half_t* kf  = (half_t*)d_ws;                             // [HOPS][B][LK][D] f16
  half_t* cf  = kf  + (size_t)HOPS * BATCH * LK * D;       // [HOPS][B][LC][D] f16
  half_t* kfT = cf  + (size_t)HOPS * BATCH * LC * D;       // [HOPS][B][D][LK] f16

  const int rows = HOPS * BATCH * (LK + LC);               // one wave per row
  dim3 g1((rows + 7) / 8);
  embed_bag_kernel<<<g1, 256, 0, stream>>>(conv_seqs, kb_arr, Ctab, Ktab, kf, cf);

  dim3 gt(LK / 32, D / 32, HOPS * BATCH);                  // 16 x 4 x 48 tiles
  transpose_kf_kernel<<<gt, 256, 0, stream>>>(kf, kfT);

  dim3 g2(LC / MTILE, BATCH);                              // 8 x 16 = 128 blocks
  memnn_attn_kernel<<<g2, 256, 0, stream>>>(kf, kfT, cf, out);
}
